// GroupQueryAttention_83786222011012
// MI455X (gfx1250) — compile-verified, matrix operand order checked by reading, not run on hardware
//
#include <hip/hip_runtime.h>
#include <hip/hip_bf16.h>

// ---------------------------------------------------------------------------
// GQA attention for MI455X (gfx1250): bf16 WMMA everywhere, f32 accumulate.
// GEMMs use async global->LDS double-buffered staging (ASYNCcnt path).
// B=1, T=2048, D=4096, H=32, HD=128, G=8, REP=4, KV_DIM=2048
// ---------------------------------------------------------------------------

#define T_LEN   2048
#define D_DIM   4096
#define H_NUM   32
#define G_NUM   8
#define HD_DIM  128
#define REP_N   4
#define KV_DIM  2048

typedef unsigned short u16;
typedef __attribute__((ext_vector_type(16))) __bf16 bf16x16;
typedef __attribute__((ext_vector_type(8)))  __bf16 bf16x8;
typedef __attribute__((ext_vector_type(8)))  float  f32x8;

union B128 { uint4 u; bf16x8 v; };

static __device__ __forceinline__ u16 f2bf(float f) {
    union { float f; unsigned u; } v; v.f = f;
    unsigned r = v.u + 0x7fffu + ((v.u >> 16) & 1u);   // round-to-nearest-even
    return (u16)(r >> 16);
}

// Load one 16x32 bf16 WMMA A/B fragment from a row-major [rows, stride] global
// buffer. Per ISA layout: lanes 0-15 -> row r, K {0..7,16..23};
// lanes 16-31 -> K {8..15,24..31}.
static __device__ __forceinline__ bf16x16
load_frag(const u16* __restrict__ base, int stride, int row0, int kcol, int lane) {
    int r  = lane & 15;
    int ks = lane >> 4;
    const u16* p = base + (size_t)(row0 + r) * stride + kcol + ks * 8;
    B128 lo, hi;
    lo.u = *(const uint4*)(p);
    hi.u = *(const uint4*)(p + 16);
    bf16x16 a;
#pragma unroll
    for (int i = 0; i < 8; ++i) { a[i] = lo.v[i]; a[i + 8] = hi.v[i]; }
    return a;
}

// Same fragment gather, but from an LDS tile laid out row-major [rows][32].
static __device__ __forceinline__ bf16x16
lds_frag(const u16* smem, unsigned base_elems, int row0, int lane) {
    int r  = lane & 15;
    int ks = lane >> 4;
    const u16* p = smem + base_elems + (unsigned)(row0 + r) * 32u + ks * 8;
    B128 lo, hi;
    lo.u = *(const uint4*)(p);          // ds_load_b128
    hi.u = *(const uint4*)(p + 16);     // ds_load_b128
    bf16x16 a;
#pragma unroll
    for (int i = 0; i < 8; ++i) { a[i] = lo.v[i]; a[i + 8] = hi.v[i]; }
    return a;
}

static __device__ __forceinline__ f32x8 f32x8_zero() {
    f32x8 z = {0.f, 0.f, 0.f, 0.f, 0.f, 0.f, 0.f, 0.f};
    return z;
}

// Async-copy a [128 rows x 32 cols] bf16 tile (8 KB) from global (row-major,
// stride ld) into LDS at byte offset lds_base. 256 threads x 2 chunks x 16 B.
// Tracked by ASYNCcnt (2 instructions per thread).
static __device__ __forceinline__ void
async_copy_tile(const u16* __restrict__ gbase, int ld, int row0, int k0,
                unsigned lds_base, int tid) {
#pragma unroll
    for (int h = 0; h < 2; ++h) {
        int c   = tid + h * 256;       // 0..511 16-byte chunks
        int row = c >> 2;
        int seg = c & 3;
        unsigned long long ga = (unsigned long long)(uintptr_t)
            (gbase + (size_t)(row0 + row) * ld + k0 + seg * 8);
        unsigned la = lds_base + (unsigned)(row * 64 + seg * 16);
        asm volatile("global_load_async_to_lds_b128 %0, %1, off"
                     :: "v"(la), "v"(ga) : "memory");
    }
}

// ------------------------- conversion kernels ------------------------------

__global__ void k_cvt_bf16(const float* __restrict__ src, u16* __restrict__ dst, int n) {
    int i = blockIdx.x * blockDim.x + threadIdx.x;
    if (i < n) dst[i] = f2bf(src[i]);
}

// src: [K, N] row-major f32  ->  dst: [N, K] row-major bf16 (transposed)
__global__ void k_cvt_transpose(const float* __restrict__ src, u16* __restrict__ dst,
                                int K, int N) {
    int i = blockIdx.x * blockDim.x + threadIdx.x;
    if (i >= K * N) return;
    int k = i / N, n = i % N;
    dst[(size_t)n * K + k] = f2bf(src[i]);
}

// KV [T, KV_DIM] -> Kb [G, T, HD]  and  Vt [G, HD, T]
__global__ void k_split_kv(const u16* __restrict__ kv, u16* __restrict__ Kb,
                           u16* __restrict__ Vt) {
    int i = blockIdx.x * blockDim.x + threadIdx.x;
    if (i >= G_NUM * T_LEN * HD_DIM) return;
    int hd = i % HD_DIM;
    int t  = (i / HD_DIM) % T_LEN;
    int g  = i / (HD_DIM * T_LEN);
    size_t row = (size_t)t * KV_DIM + g * 2 * HD_DIM;
    Kb[i] = kv[row + hd];
    Vt[(size_t)g * HD_DIM * T_LEN + (size_t)hd * T_LEN + t] = kv[row + HD_DIM + hd];
}

// ------------------------------ WMMA GEMM ----------------------------------
// C[M,N] = A[M,K] * B[K,N], with B pre-transposed to Bt[N,K] (both bf16).
// Block = 256 threads (8 waves). Block tile 128x128; wave tile 32x64.
// A/B 128x32 tiles are staged to LDS with async copies, double-buffered:
//   LDS layout: buf b at byte b*16384: [A tile 8KB][B tile 8KB]  (32 KB total)
// mode 0: bf16 output.  mode 1: f32 output + bias.

__global__ __launch_bounds__(256)
void k_gemm_bf16(const u16* __restrict__ A, const u16* __restrict__ Bt,
                 u16* __restrict__ Cbf, float* __restrict__ Cf,
                 const float* __restrict__ bias,
                 int M, int N, int K, int lda, int ldb, int ldc, int mode) {
    extern __shared__ __attribute__((aligned(16))) u16 smem[];
    int tid  = threadIdx.x;
    int lane = tid & 31;
    int w    = tid >> 5;
    int wm   = w & 3;         // 4 waves along M
    int wn   = w >> 2;        // 2 waves along N
    int m0   = blockIdx.y * 128;
    int n0   = blockIdx.x * 128;
    (void)M; (void)N;

    f32x8 acc[2][4];
#pragma unroll
    for (int i = 0; i < 2; ++i)
#pragma unroll
        for (int j = 0; j < 4; ++j) acc[i][j] = f32x8_zero();

    const int nk = K >> 5;

    // prologue: stage k-step 0 into buffer 0
    async_copy_tile(A,  lda, m0, 0, 0u,    tid);
    async_copy_tile(Bt, ldb, n0, 0, 8192u, tid);

    for (int ks = 0; ks < nk; ++ks) {
        int cur = ks & 1;
        unsigned curb = (unsigned)cur * 16384u;

        if (ks + 1 < nk) {
            unsigned nxtb = (unsigned)(1 - cur) * 16384u;
            async_copy_tile(A,  lda, m0, (ks + 1) * 32, nxtb,         tid);
            async_copy_tile(Bt, ldb, n0, (ks + 1) * 32, nxtb + 8192u, tid);
            // current buffer's 4 copies (issued last iter) complete in order;
            // allow the 4 just-issued to stay in flight.
            asm volatile("s_wait_asynccnt 0x4" ::: "memory");
        } else {
            asm volatile("s_wait_asynccnt 0x0" ::: "memory");
        }
        __syncthreads();   // all waves' copies into cur are visible

        unsigned aE = curb >> 1;                // elem offsets
        unsigned bE = (curb + 8192u) >> 1;
        bf16x16 a0 = lds_frag(smem, aE, wm * 32,      lane);
        bf16x16 a1 = lds_frag(smem, aE, wm * 32 + 16, lane);
#pragma unroll
        for (int j = 0; j < 4; ++j) {
            bf16x16 b = lds_frag(smem, bE, wn * 64 + j * 16, lane);
            acc[0][j] = __builtin_amdgcn_wmma_f32_16x16x32_bf16(
                false, a0, false, b, (short)0, acc[0][j], false, false);
            acc[1][j] = __builtin_amdgcn_wmma_f32_16x16x32_bf16(
                false, a1, false, b, (short)0, acc[1][j], false, false);
        }
        __syncthreads();   // done reading cur before it is overwritten
    }

    // Epilogue. C/D layout: VGPR e holds row e (lanes 0-15) / row e+8 (lanes 16-31),
    // column = lane & 15.
    int cn = lane & 15, ks2 = lane >> 4;
#pragma unroll
    for (int i = 0; i < 2; ++i) {
        int rbase = m0 + wm * 32 + i * 16 + ks2 * 8;
#pragma unroll
        for (int j = 0; j < 4; ++j) {
            int col = n0 + wn * 64 + j * 16 + cn;
#pragma unroll
            for (int e = 0; e < 8; ++e) {
                size_t idx = (size_t)(rbase + e) * ldc + col;
                if (mode == 0) Cbf[idx] = f2bf(acc[i][j][e]);
                else           Cf[idx]  = acc[i][j][e] + bias[col];
            }
        }
    }
}

// --------------------------- flash attention -------------------------------
// Grid: (T/128, H). Block = 256 thr = 8 waves; wave w owns 16 query rows.
// S = Q*K^T (8 wmma / 32-key tile), online softmax, P via per-wave LDS
// transpose, O += P*V (8 wmma / tile). Output bf16 ctx[T, D].

__global__ __launch_bounds__(256)
void k_flash_attn(const u16* __restrict__ Q,   // [T, D] bf16
                  const u16* __restrict__ Kb,  // [G, T, HD]
                  const u16* __restrict__ Vt,  // [G, HD, T]
                  u16* __restrict__ Ctx) {     // [T, D]
    __shared__ __attribute__((aligned(16))) u16 plds[8][16 * 32];

    int lane = threadIdx.x & 31;
    int w    = threadIdx.x >> 5;
    int q0   = blockIdx.x * 128 + w * 16;
    int h    = blockIdx.y;
    int g    = h / REP_N;

    const u16* Qh = Q  + (size_t)h * HD_DIM;                 // row stride D
    const u16* Kg = Kb + (size_t)g * T_LEN * HD_DIM;         // [T, HD]
    const u16* Vg = Vt + (size_t)g * HD_DIM * T_LEN;         // [HD, T]

    // Q fragments for all 4 K-chunks of HD=128 (kept in registers).
    bf16x16 aq[4];
#pragma unroll
    for (int kc = 0; kc < 4; ++kc)
        aq[kc] = load_frag(Qh, D_DIM, q0, kc * 32, lane);

    f32x8 O[8];
#pragma unroll
    for (int f = 0; f < 8; ++f) O[f] = f32x8_zero();

    float mrow[8], lrow[8];
#pragma unroll
    for (int j = 0; j < 8; ++j) { mrow[j] = -3.0e38f; lrow[j] = 0.f; }

    const float sscale = 0.08838834764831845f * 1.4426950408889634f; // 1/sqrt(HD) * log2(e)
    int cn = lane & 15, ks = lane >> 4;

    int kend = q0 + 16;                       // causal: keys <= last query row
    for (int key0 = 0; key0 < kend; key0 += 32) {
        f32x8 S[2];
        S[0] = f32x8_zero(); S[1] = f32x8_zero();
#pragma unroll
        for (int f = 0; f < 2; ++f)
#pragma unroll
            for (int kc = 0; kc < 4; ++kc) {
                bf16x16 bk = load_frag(Kg, HD_DIM, key0 + f * 16, kc * 32, lane);
                S[f] = __builtin_amdgcn_wmma_f32_16x16x32_bf16(
                    false, aq[kc], false, bk, (short)0, S[f], false, false);
            }

        // online softmax update, row j (+8 for upper half) per lane
#pragma unroll
        for (int j = 0; j < 8; ++j) {
            int row = q0 + j + ks * 8;
            float s0 = (key0 + cn      <= row) ? S[0][j] * sscale : -1.0e30f;
            float s1 = (key0 + 16 + cn <= row) ? S[1][j] * sscale : -1.0e30f;
            float rm = fmaxf(s0, s1);
#pragma unroll
            for (int d = 1; d < 16; d <<= 1) rm = fmaxf(rm, __shfl_xor(rm, d));
            float mn    = fmaxf(mrow[j], rm);
            float alpha = exp2f(mrow[j] - mn);
            float p0    = exp2f(s0 - mn);
            float p1    = exp2f(s1 - mn);
            float rs    = p0 + p1;
#pragma unroll
            for (int d = 1; d < 16; d <<= 1) rs += __shfl_xor(rs, d);
            lrow[j] = lrow[j] * alpha + rs;
            mrow[j] = mn;
#pragma unroll
            for (int f = 0; f < 8; ++f) O[f][j] *= alpha;
            // P in C-layout -> LDS (row-major 16x32 bf16 per wave)
            plds[w][(j + ks * 8) * 32 + cn]      = f2bf(p0);
            plds[w][(j + ks * 8) * 32 + 16 + cn] = f2bf(p1);
        }
        asm volatile("s_wait_dscnt 0x0" ::: "memory");   // same-wave LDS RAW

        // Re-read P as a 16x32 A-fragment (ds_load_b128 x2)
        bf16x16 ap;
        {
            const u16* p = &plds[w][(lane & 15) * 32 + (lane >> 4) * 8];
            B128 lo, hi;
            lo.u = *(const uint4*)(p);
            hi.u = *(const uint4*)(p + 16);
#pragma unroll
            for (int i = 0; i < 8; ++i) { ap[i] = lo.v[i]; ap[i + 8] = hi.v[i]; }
        }

        // O[:, f*16 .. f*16+15] += P (16x32 keys) * V (32 keys x 16 hd)
#pragma unroll
        for (int f = 0; f < 8; ++f) {
            bf16x16 bv = load_frag(Vg, T_LEN, f * 16, key0, lane);
            O[f] = __builtin_amdgcn_wmma_f32_16x16x32_bf16(
                false, ap, false, bv, (short)0, O[f], false, false);
        }
    }

    // epilogue: normalize and store ctx rows
#pragma unroll
    for (int f = 0; f < 8; ++f) {
        int col = h * HD_DIM + f * 16 + cn;
#pragma unroll
        for (int j = 0; j < 8; ++j) {
            int row = q0 + j + ks * 8;
            Ctx[(size_t)row * D_DIM + col] = f2bf(O[f][j] / lrow[j]);
        }
    }
}

// ------------------------------- launch ------------------------------------

extern "C" void kernel_launch(void* const* d_in, const int* in_sizes, int n_in,
                              void* d_out, int out_size, void* d_ws, size_t ws_size,
                              hipStream_t stream) {
    (void)in_sizes; (void)n_in; (void)out_size; (void)ws_size;
    const float* x   = (const float*)d_in[0];
    const float* Wq  = (const float*)d_in[1];
    const float* Wkv = (const float*)d_in[2];
    const float* Wo  = (const float*)d_in[3];
    const float* bo  = (const float*)d_in[4];
    float* out = (float*)d_out;

    // workspace layout (bf16 buffers)
    char* ws = (char*)d_ws;
    size_t off = 0;
    u16* xbf  = (u16*)(ws + off); off += (size_t)T_LEN * D_DIM * 2;      // 16 MB
    u16* WqT  = (u16*)(ws + off); off += (size_t)D_DIM * D_DIM * 2;      // 32 MB
    u16* WkvT = (u16*)(ws + off); off += (size_t)D_DIM * KV_DIM * 2;     // 16 MB
    u16* WoT  = (u16*)(ws + off); off += (size_t)D_DIM * D_DIM * 2;      // 32 MB
    u16* Qbf  = (u16*)(ws + off); off += (size_t)T_LEN * D_DIM * 2;      // 16 MB
    u16* KVbf = (u16*)(ws + off); off += (size_t)T_LEN * KV_DIM * 2;     //  8 MB
    u16* Kbf  = (u16*)(ws + off); off += (size_t)G_NUM * T_LEN * HD_DIM * 2; // 4 MB
    u16* Vtb  = (u16*)(ws + off); off += (size_t)G_NUM * HD_DIM * T_LEN * 2; // 4 MB
    u16* Ctxb = (u16*)(ws + off); off += (size_t)T_LEN * D_DIM * 2;      // 16 MB

    const int TPB = 256;
    const size_t GEMM_LDS = 32768;   // 2 buffers x (8KB A + 8KB B)

    // 1) conversions
    {
        int n = T_LEN * D_DIM;
        k_cvt_bf16<<<(n + TPB - 1) / TPB, TPB, 0, stream>>>(x, xbf, n);
    }
    k_cvt_transpose<<<(D_DIM * D_DIM + TPB - 1) / TPB, TPB, 0, stream>>>(Wq,  WqT,  D_DIM, D_DIM);
    k_cvt_transpose<<<(D_DIM * KV_DIM + TPB - 1) / TPB, TPB, 0, stream>>>(Wkv, WkvT, D_DIM, KV_DIM);
    k_cvt_transpose<<<(D_DIM * D_DIM + TPB - 1) / TPB, TPB, 0, stream>>>(Wo,  WoT,  D_DIM, D_DIM);

    // 2) Q = x * Wq  (bf16 out)
    k_gemm_bf16<<<dim3(D_DIM / 128, T_LEN / 128), TPB, GEMM_LDS, stream>>>(
        xbf, WqT, Qbf, nullptr, nullptr,
        T_LEN, D_DIM, D_DIM, D_DIM, D_DIM, D_DIM, 0);

    // 3) KV = x * Wkv (bf16 out)
    k_gemm_bf16<<<dim3(KV_DIM / 128, T_LEN / 128), TPB, GEMM_LDS, stream>>>(
        xbf, WkvT, KVbf, nullptr, nullptr,
        T_LEN, KV_DIM, D_DIM, D_DIM, D_DIM, KV_DIM, 0);

    // 4) split KV into K[g][t][hd] and V^T[g][hd][t]
    {
        int n = G_NUM * T_LEN * HD_DIM;
        k_split_kv<<<(n + TPB - 1) / TPB, TPB, 0, stream>>>(KVbf, Kbf, Vtb);
    }

    // 5) flash attention -> ctx bf16
    k_flash_attn<<<dim3(T_LEN / 128, H_NUM), TPB, 0, stream>>>(Qbf, Kbf, Vtb, Ctxb);

    // 6) out = ctx * Wo + bo  (f32 out)
    k_gemm_bf16<<<dim3(D_DIM / 128, T_LEN / 128), TPB, GEMM_LDS, stream>>>(
        Ctxb, WoT, nullptr, out, bo,
        T_LEN, D_DIM, D_DIM, D_DIM, D_DIM, D_DIM, 1);
}
